// Aresblock_51092930953592
// MI455X (gfx1250) — 8-bit matrix operand order repaired and checked by reading, compiled, not run on hardware
//
#include <hip/hip_runtime.h>
#include <hip/hip_bf16.h>


typedef int v8i __attribute__((ext_vector_type(8)));

#define HW    3136
#define NB    32
#define CCH   256
#define HALF  128
#define MTILE 64
#define TPB   49
#define EPSV  1e-5f

__global__ void wscale_kernel(const float* __restrict__ w, int ksize2,
                              float* __restrict__ scales) {
  const int o = blockIdx.x;
  const int KE = 128 * ksize2;
  __shared__ float red[256];
  float s = 0.f;
  const float* p = w + (size_t)o * KE;
  for (int i = threadIdx.x; i < KE; i += blockDim.x) s += fabsf(p[i]);
  red[threadIdx.x] = s;
  __syncthreads();
  for (int st = 128; st > 0; st >>= 1) {
    if (threadIdx.x < st) red[threadIdx.x] += red[threadIdx.x + st];
    __syncthreads();
  }
  if (threadIdx.x == 0) scales[o] = red[0] / (float)KE;
}

__global__ void pack_w_kernel(const float* __restrict__ w, int ksteps,
                              int ksize2, int* __restrict__ packed) {
  int blk = blockIdx.x;
  int ks = blk % ksteps;
  int nt = (blk / ksteps) & 3;
  int g  = blk / (ksteps * 4);
  int t = threadIdx.x;
  int v = t & 7;
  int lane = t >> 3;
  int n = lane & 15;
  int o = g * 64 + nt * 16 + n;
  int kbase = ((v >= 4) ? 32 : 0) + (lane >> 4) * 16 + (v & 3) * 4;
  unsigned d = 0;
  for (int j = 0; j < 4; ++j) {
    int kk = ks * 64 + kbase + j;
    int ci = kk / ksize2;
    int r  = kk - ci * ksize2;
    float wv = w[((size_t)o * 128 + ci) * ksize2 + r];
    int sgn = (wv > 0.f) ? 1 : ((wv < 0.f) ? -1 : 0);
    d |= ((unsigned)(sgn & 0xff)) << (8 * j);
  }
  packed[(((size_t)(g * 4 + nt) * ksteps + ks) * 32 + lane) * 8 + v] = (int)d;
}

__global__ void binact_shuffle_kernel(const float* __restrict__ x,
                                      const float* __restrict__ b1,
                                      signed char* __restrict__ s8a) {
  size_t i = (size_t)blockIdx.x * blockDim.x + threadIdx.x;
  int hw = (int)(i % HW);
  size_t tt = i / HW;
  int c = (int)(tt % CCH);
  int b = (int)(tt / CCH);
  int oc = (c & 1) * 128 + (c >> 1);
  float v = x[(((size_t)b * CCH + oc) * HW) + hw] + b1[c];
  s8a[i] = (v > 0.f) ? 1 : ((v < 0.f) ? -1 : 0);
}

template <int KS3>
__global__ void bin_conv_wmma(const signed char* __restrict__ s8,
                              const int* __restrict__ packB,
                              const float* __restrict__ scales,
                              float* __restrict__ yraw) {
  constexpr int KSTEPS = KS3 ? 18 : 2;
  __shared__ signed char lA[MTILE * 64];

  const int tile = blockIdx.x;
  const int g    = blockIdx.y;
  const int b    = tile / TPB;
  const int hw0  = (tile % TPB) * MTILE;
  const int t    = threadIdx.x;
  const int lane = t & 31;
  const int nt   = t >> 5;
  const int Mm   = lane & 15;

  const int fm     = t >> 1;
  const int fkoff  = (t & 1) * 32;
  const int fhw    = hw0 + fm;
  const int fh     = fhw / 56;
  const int fw     = fhw - fh * 56;
  const signed char* splane = s8 + ((size_t)b * CCH + g * 128) * HW;

  v8i acc0 = {}, acc1 = {}, acc2 = {}, acc3 = {};

  for (int ks = 0; ks < KSTEPS; ++ks) {
#pragma unroll
    for (int d = 0; d < 8; ++d) {
      unsigned pk = 0;
#pragma unroll
      for (int j = 0; j < 4; ++j) {
        int k  = fkoff + d * 4 + j;
        int kk = ks * 64 + k;
        signed char sv;
        if (KS3) {
          int ci = kk / 9;
          int r  = kk - ci * 9;
          int kh = r / 3;
          int kw = r - kh * 3;
          int hi = fh + kh - 1, wi = fw + kw - 1;
          sv = ((unsigned)hi < 56u && (unsigned)wi < 56u)
                   ? splane[(size_t)ci * HW + hi * 56 + wi]
                   : (signed char)0;
        } else {
          sv = splane[(size_t)kk * HW + fhw];
        }
        pk |= ((unsigned)(unsigned char)sv) << (8 * j);
      }
      *(unsigned*)&lA[fm * 64 + fkoff + d * 4] = pk;
    }
    __syncthreads();

    const int* bp =
        packB + ((((size_t)g * 4 + nt) * KSTEPS + ks) * 32 + lane) * 8;
    v8i bfrag = *(const v8i*)bp;
    if (ks + 1 < KSTEPS) __builtin_prefetch(bp + 32 * 8, 0, 3);

#pragma unroll
    for (int mt = 0; mt < 4; ++mt) {
      v8i afrag;
#pragma unroll
      for (int vv = 0; vv < 8; ++vv) {
        int kb = (vv >> 2) * 32 + (lane >> 4) * 16 + (vv & 3) * 4;
        afrag[vv] = *(const int*)&lA[(mt * 16 + Mm) * 64 + kb];
      }
      v8i* accp = (mt == 0) ? &acc0 : (mt == 1) ? &acc1 : (mt == 2) ? &acc2
                                                                    : &acc3;
      *accp = __builtin_amdgcn_wmma_i32_16x16x64_iu8(
           true, afrag,  true, bfrag, *accp,
           false,  false);
    }
    __syncthreads();
  }

  const int cg = g * 64 + nt * 16 + Mm;
  const float s = scales[cg];
  float* yout = yraw + (((size_t)b * HALF + cg) * HW) + hw0;
#pragma unroll
  for (int mt = 0; mt < 4; ++mt) {
    v8i acc = (mt == 0) ? acc0 : (mt == 1) ? acc1 : (mt == 2) ? acc2 : acc3;
#pragma unroll
    for (int r = 0; r < 8; ++r) {
      int M = mt * 16 + r + 8 * (lane >> 4);
      yout[M] = (float)acc[r] * s;
    }
  }
}

__global__ void bn_stats_kernel(const float* __restrict__ yraw,
                                float* __restrict__ stats) {
  const int c = blockIdx.x;
  __shared__ float ssum[256], ssq[256];
  float s = 0.f, q = 0.f;
  for (int b = 0; b < NB; ++b) {
    const float* p = yraw + ((size_t)b * HALF + c) * HW;
    for (int i = threadIdx.x; i < HW; i += blockDim.x) {
      float v = p[i];
      s += v;
      q += v * v;
    }
  }
  ssum[threadIdx.x] = s;
  ssq[threadIdx.x] = q;
  __syncthreads();
  for (int st = 128; st > 0; st >>= 1) {
    if (threadIdx.x < st) {
      ssum[threadIdx.x] += ssum[threadIdx.x + st];
      ssq[threadIdx.x] += ssq[threadIdx.x + st];
    }
    __syncthreads();
  }
  if (threadIdx.x == 0) {
    float N = (float)NB * (float)HW;
    float mu = ssum[0] / N;
    float var = ssq[0] / N - mu * mu;
    stats[c] = mu;
    stats[HALF + c] = rsqrtf(var + EPSV);
  }
}

__global__ void ew_mid_kernel(const float* __restrict__ x,
                              const float* __restrict__ yraw,
                              const float* __restrict__ stats,
                              const float* __restrict__ g1,
                              const float* __restrict__ be1,
                              const float* __restrict__ b2_1,
                              const float* __restrict__ a2,
                              const float* __restrict__ b2_2,
                              const float* __restrict__ b3_1,
                              float* __restrict__ x2,
                              signed char* __restrict__ s8) {
  size_t i = (size_t)blockIdx.x * blockDim.x + threadIdx.x;
  int hw = (int)(i % HW);
  size_t tt = i / HW;
  int c = (int)(tt % CCH);
  int b = (int)(tt / CCH);
  int oc = (c & 1) * 128 + (c >> 1);
  float xsh = x[(((size_t)b * CCH + oc) * HW) + hw];
  float v;
  if (c < HALF) {
    float y = yraw[(((size_t)b * HALF + c) * HW) + hw];
    y = (y - stats[c]) * stats[HALF + c] * g1[c] + be1[c];
    v = y + xsh;
  } else {
    v = xsh;
  }
  float u = v + b2_1[c];
  u = (u >= 0.f) ? u : a2[c] * u;
  u = u + b2_2[c] + b3_1[c];
  x2[i] = u;
  s8[i] = (u > 0.f) ? 1 : ((u < 0.f) ? -1 : 0);
}

__global__ void ew_out_kernel(const float* __restrict__ x2,
                              const float* __restrict__ yraw,
                              const float* __restrict__ stats,
                              const float* __restrict__ g3,
                              const float* __restrict__ be3,
                              const float* __restrict__ b4_1,
                              const float* __restrict__ a4,
                              const float* __restrict__ b4_2,
                              float* __restrict__ out) {
  size_t i = (size_t)blockIdx.x * blockDim.x + threadIdx.x;
  int hw = (int)(i % HW);
  size_t tt = i / HW;
  int c = (int)(tt % CCH);
  int b = (int)(tt / CCH);
  float v;
  if (c < HALF) {
    float y = yraw[(((size_t)b * HALF + c) * HW) + hw];
    y = (y - stats[c]) * stats[HALF + c] * g3[c] + be3[c];
    v = y + x2[i];
  } else {
    v = x2[i];
  }
  float u = v + b4_1[c];
  u = (u >= 0.f) ? u : a4[c] * u;
  out[i] = u + b4_2[c];
}

extern "C" void kernel_launch(void* const* d_in, const int* in_sizes, int n_in,
                              void* d_out, int out_size, void* d_ws,
                              size_t ws_size, hipStream_t stream) {
  const float* x    = (const float*)d_in[0];
  const float* w3   = (const float*)d_in[1];
  const float* w1   = (const float*)d_in[2];
  const float* b1   = (const float*)d_in[3];
  const float* g1   = (const float*)d_in[4];
  const float* be1  = (const float*)d_in[5];
  const float* b2_1 = (const float*)d_in[6];
  const float* a2   = (const float*)d_in[7];
  const float* b2_2 = (const float*)d_in[8];
  const float* b3_1 = (const float*)d_in[9];
  const float* g3   = (const float*)d_in[10];
  const float* be3  = (const float*)d_in[11];
  const float* b4_1 = (const float*)d_in[12];
  const float* a4   = (const float*)d_in[13];
  const float* b4_2 = (const float*)d_in[14];

  char* ws = (char*)d_ws;
  int*   packB3 = (int*)(ws + 0);
  int*   packB1 = (int*)(ws + 160 * 1024);
  float* sc3    = (float*)(ws + 192 * 1024);
  float* sc1    = (float*)(ws + 196 * 1024);
  float* st1    = (float*)(ws + 200 * 1024);
  float* st3    = (float*)(ws + 204 * 1024);
  const size_t yraw_off = 1ull << 20;
  const size_t yraw_sz  = (size_t)NB * HALF * HW * 4;
  const size_t x2_off   = yraw_off + yraw_sz;
  const size_t x2_sz    = (size_t)NB * CCH * HW * 4;
  const size_t s8_sz    = (size_t)NB * CCH * HW;
  const size_t s8b_off  = x2_off + x2_sz;
  const size_t s8a_off  = s8b_off + s8_sz;
  float* yraw = (float*)(ws + yraw_off);
  float* x2   = (float*)(ws + x2_off);
  signed char* s8b = (signed char*)(ws + s8b_off);
  signed char* s8a = (signed char*)(ws + s8a_off);

  const size_t tot = (size_t)NB * CCH * HW;

  wscale_kernel<<<128, 256, 0, stream>>>(w3, 9, sc3);
  wscale_kernel<<<128, 256, 0, stream>>>(w1, 1, sc1);
  pack_w_kernel<<<2 * 4 * 18, 256, 0, stream>>>(w3, 18, 9, packB3);
  pack_w_kernel<<<2 * 4 * 2, 256, 0, stream>>>(w1, 2, 1, packB1);
  binact_shuffle_kernel<<<(unsigned)(tot / 256), 256, 0, stream>>>(x, b1, s8a);

  dim3 cgrid(NB * TPB, 2);

  bin_conv_wmma<1><<<cgrid, 128, 0, stream>>>(s8a, packB3, sc3, yraw);
  bn_stats_kernel<<<128, 256, 0, stream>>>(yraw, st1);

  ew_mid_kernel<<<(unsigned)(tot / 256), 256, 0, stream>>>(
      x, yraw, st1, g1, be1, b2_1, a2, b2_2, b3_1, x2, s8b);

  bin_conv_wmma<0><<<cgrid, 128, 0, stream>>>(s8b, packB1, sc1, yraw);
  bn_stats_kernel<<<128, 256, 0, stream>>>(yraw, st3);

  ew_out_kernel<<<(unsigned)(tot / 256), 256, 0, stream>>>(
      x2, yraw, st3, g3, be3, b4_1, a4, b4_2, (float*)d_out);
}
